// LiveNet_20504173871714
// MI455X (gfx1250) — compile-verified, matrix operand order checked by reading, not run on hardware
//
#include <hip/hip_runtime.h>

#define N_IN  2048
#define N_MID 8192
#define N_OUT 2048
#define BATCH 4096

typedef __attribute__((ext_vector_type(16))) __bf16 v16bf;
typedef __attribute__((ext_vector_type(8)))  __bf16 v8bf;
typedef __attribute__((ext_vector_type(8)))  float  v8f;
typedef __attribute__((ext_vector_type(4)))  float  v4f;

// ---------------------------------------------------------------------------
// One-shot f32 -> bf16 conversion (memory-bound). 8 elements per thread;
// n must be a multiple of 8.
// ---------------------------------------------------------------------------
__global__ __launch_bounds__(256)
void cvt_f32_bf16(const float* __restrict__ in, __bf16* __restrict__ out, long n) {
  long i = ((long)blockIdx.x * 256 + threadIdx.x) * 8;
  if (i >= n) return;
  v4f a = *(const v4f*)(in + i);
  v4f b = *(const v4f*)(in + i + 4);
  v8bf o;
#pragma unroll
  for (int j = 0; j < 4; ++j) {
    o[j]     = (__bf16)a[j];
    o[j + 4] = (__bf16)b[j];
  }
  *(v8bf*)(out + i) = o;
}

// ---------------------------------------------------------------------------
// Load one 16x32 bf16 WMMA operand fragment from a row-major [rows, ld] bf16
// matrix (K contiguous). CDNA5 layout: lanes 0-15 hold row=lane, K klo..klo+7
// in v[0..3] and K klo+16..klo+23 in v[4..7]; lanes 16-31 use klo=8.
// Two 16B loads per lane; WGP-scope prefetch 4 K-steps ahead (locality 3 so
// the line is pulled into WGP$, which absorbs the intra-block re-reads).
// ---------------------------------------------------------------------------
__device__ __forceinline__ v16bf load_frag(const __bf16* __restrict__ base,
                                           int row0, int ld, int k, int lane) {
  const int r   = row0 + (lane & 15);
  const int klo = (lane >> 4) << 3;          // 0 or 8
  const __bf16* p = base + (size_t)r * ld + k + klo;
  v8bf lo = *(const v8bf*)(p);
  v8bf hi = *(const v8bf*)(p + 16);
  __builtin_prefetch(p + 128, 0, 3);         // WGP-scope global_prefetch_b8
  return __builtin_shufflevector(lo, hi, 0, 1, 2, 3, 4, 5, 6, 7,
                                 8, 9, 10, 11, 12, 13, 14, 15);
}

// ---------------------------------------------------------------------------
// C[M,N] = relu(A[M,K] @ B[N,K]^T + bias[N]), A/B bf16, acc f32.
// Block = 256 threads = 8 waves in a 2x4 grid; wave tile 64x64 (4x4 WMMA
// fragments), block tile 128x256. 16 v_wmma per K-step of 32 per wave.
// ---------------------------------------------------------------------------
template <bool OUT_BF16>
__global__ __launch_bounds__(256)
void gemm_bias_relu(const __bf16* __restrict__ A, const __bf16* __restrict__ B,
                    const float* __restrict__ bias, void* __restrict__ Cv,
                    int M, int N, int K) {
  const int lane = threadIdx.x & 31;
  const int wave = threadIdx.x >> 5;
  const int wm = wave >> 2;                       // 0..1 (M dir)
  const int wn = wave & 3;                        // 0..3 (N dir)
  const int m0 = blockIdx.y * 128 + wm * 64;
  const int n0 = blockIdx.x * 256 + wn * 64;

  v8f acc[4][4] = {};

  for (int k = 0; k < K; k += 32) {
    v16bf a[4], b[4];
#pragma unroll
    for (int i = 0; i < 4; ++i)
      a[i] = load_frag(A, m0 + 16 * i, K, k, lane);
#pragma unroll
    for (int j = 0; j < 4; ++j)
      b[j] = load_frag(B, n0 + 16 * j, K, k, lane);

#pragma unroll
    for (int i = 0; i < 4; ++i)
#pragma unroll
      for (int j = 0; j < 4; ++j)
        acc[i][j] = __builtin_amdgcn_wmma_f32_16x16x32_bf16(
            false, a[i], false, b[j], (short)0, acc[i][j], false, false);
  }

  // C/D layout: lane(0-15) = N col; lanes 16-31 are rows +8; VGPR v = row + v.
  const int nn = lane & 15;
  const int mb = (lane >> 4) << 3;
#pragma unroll
  for (int i = 0; i < 4; ++i) {
#pragma unroll
    for (int j = 0; j < 4; ++j) {
      const int col = n0 + 16 * j + nn;
      const float bv = bias[col];
#pragma unroll
      for (int v = 0; v < 8; ++v) {
        const int row = m0 + 16 * i + mb + v;
        float r = acc[i][j][v] + bv;
        r = r > 0.f ? r : 0.f;
        if constexpr (OUT_BF16)
          ((__bf16*)Cv)[(size_t)row * N + col] = (__bf16)r;
        else
          ((float*)Cv)[(size_t)row * N + col] = r;
      }
    }
  }
}

extern "C" void kernel_launch(void* const* d_in, const int* in_sizes, int n_in,
                              void* d_out, int out_size, void* d_ws, size_t ws_size,
                              hipStream_t stream) {
  (void)in_sizes; (void)n_in; (void)out_size; (void)ws_size;
  const float* x  = (const float*)d_in[0];
  const float* W1 = (const float*)d_in[1];
  const float* b1 = (const float*)d_in[2];
  const float* W2 = (const float*)d_in[3];
  const float* b2 = (const float*)d_in[4];

  // Workspace layout (bf16):
  //   hidden [BATCH, N_MID]  64 MB
  //   x_bf   [BATCH, N_IN]   16 MB
  //   W1_bf  [N_MID, N_IN]   32 MB
  //   W2_bf  [N_OUT, N_MID]  32 MB
  char* ws = (char*)d_ws;
  __bf16* hidden = (__bf16*)ws;
  __bf16* xb     = (__bf16*)(ws + (size_t)BATCH * N_MID * 2);
  __bf16* W1b    = xb  + (size_t)BATCH * N_IN;
  __bf16* W2b    = W1b + (size_t)N_MID * N_IN;

  dim3 blk(256, 1, 1);

  const long n_x  = (long)BATCH * N_IN;    // 8M
  const long n_w1 = (long)N_MID * N_IN;    // 16M
  const long n_w2 = (long)N_OUT * N_MID;   // 16M
  cvt_f32_bf16<<<dim3((unsigned)(n_x  / (8 * 256))), blk, 0, stream>>>(x,  xb,  n_x);
  cvt_f32_bf16<<<dim3((unsigned)(n_w1 / (8 * 256))), blk, 0, stream>>>(W1, W1b, n_w1);
  cvt_f32_bf16<<<dim3((unsigned)(n_w2 / (8 * 256))), blk, 0, stream>>>(W2, W2b, n_w2);

  dim3 g1(N_MID / 256, BATCH / 128, 1);    // 32 x 32 blocks
  gemm_bias_relu<true><<<g1, blk, 0, stream>>>(xb, W1b, b1, hidden,
                                               BATCH, N_MID, N_IN);
  dim3 g2(N_OUT / 256, BATCH / 128, 1);    // 8 x 32 blocks
  gemm_bias_relu<false><<<g2, blk, 0, stream>>>(hidden, W2b, b2, d_out,
                                                BATCH, N_OUT, N_MID);
}
